// MultiwayFusionLayer_30219389894938
// MI455X (gfx1250) — compile-verified, matrix operand order checked by reading, not run on hardware
//
#include <hip/hip_runtime.h>
#include <hip/hip_bf16.h>
#include <math.h>

// ---------------------------------------------------------------------------
// MultiwayFusionLayer for MI455X (gfx1250, wave32, WMMA)
// All GEMMs use v_wmma_f32_16x16x32_bf16 (bf16 in, f32 accumulate).
// Wave tile 64x64 (16 WMMA per 16 b128 loads); block = 8 waves = 128x256.
// GEMM epilogue writes bf16 directly when the consumer is another GEMM.
// ---------------------------------------------------------------------------

typedef __bf16 bf16;
typedef __attribute__((ext_vector_type(16))) __bf16 v16bf;
typedef __attribute__((ext_vector_type(8)))  __bf16 v8bf;
typedef __attribute__((ext_vector_type(8)))  float  v8f;

constexpr int cB = 16, cVP = 196, cTL = 316, cVD = 1024, cH = 768;
constexpr int cNH = 8, cHD = 96, cFF = 3072, cL = 6, cS = 512;

// ---------------- fragment load (A and B operands share the pattern) -------
// lane l holds row/col (l&15); K elements {base..base+7, base+16..base+23},
// base = (l>=16) ? 8 : 0  ==> two aligned 16B loads.
static __device__ __forceinline__ v16bf load_frag(const bf16* p) {
  union { v16bf v; v8bf h[2]; } f;
  f.h[0] = *(const v8bf*)(p);
  f.h[1] = *(const v8bf*)(p + 16);
  return f.v;
}

// ---------------- wave32 / block reductions --------------------------------
static __device__ __forceinline__ float waveRedSum(float v) {
#pragma unroll
  for (int o = 16; o > 0; o >>= 1) v += __shfl_xor(v, o, 32);
  return v;
}
static __device__ __forceinline__ float waveRedMax(float v) {
#pragma unroll
  for (int o = 16; o > 0; o >>= 1) v = fmaxf(v, __shfl_xor(v, o, 32));
  return v;
}
static __device__ __forceinline__ float blockRedSum(float v, float* sm) {
  __syncthreads();
  v = waveRedSum(v);
  if ((threadIdx.x & 31) == 0) sm[threadIdx.x >> 5] = v;
  __syncthreads();
  if (threadIdx.x == 0) {
    float t = 0.f;
    int nw = blockDim.x >> 5;
    for (int i = 0; i < nw; ++i) t += sm[i];
    sm[8] = t;
  }
  __syncthreads();
  return sm[8];
}
static __device__ __forceinline__ float blockRedMax(float v, float* sm) {
  __syncthreads();
  v = waveRedMax(v);
  if ((threadIdx.x & 31) == 0) sm[threadIdx.x >> 5] = v;
  __syncthreads();
  if (threadIdx.x == 0) {
    float t = -3.402823466e38f;
    int nw = blockDim.x >> 5;
    for (int i = 0; i < nw; ++i) t = fmaxf(t, sm[i]);
    sm[8] = t;
  }
  __syncthreads();
  return sm[8];
}

// ---------------- generic batched WMMA GEMM:  C = A * W^T  -----------------
// A: [batch][M,K] bf16 (row stride lda), W: [batch][N,K] bf16 (row stride ldw)
// C offset = (z/cDiv)*cS1 + (z%cDiv)*cS2; stores bf16 to Ch if non-null,
// else fp32 to Cf.
// mode: 0=none, 1=+bias[n], 2=gelu_exact(x+bias[n]), 3=x*scale + key-mask.
// Block = 8 waves arranged 2(M) x 4(N); wave tile 64x64; block tile 128x256.
__global__ __launch_bounds__(256)
void gemm_bf16_wmma(const bf16* __restrict__ Ab, const bf16* __restrict__ Wb,
                    float* __restrict__ Cf, bf16* __restrict__ Ch,
                    int M, int N, int K, int lda, int ldw, int ldc,
                    long long sA, long long sW,
                    int cDiv, long long cS1, long long cS2,
                    const float* __restrict__ bias, int mode, float scale,
                    const unsigned char* __restrict__ mask)
{
  const int bidx = blockIdx.z;
  const bf16* A = Ab + (long long)bidx * sA;
  const bf16* W = Wb + (long long)bidx * sW;
  const long long cOff = (long long)(bidx / cDiv) * cS1 + (long long)(bidx % cDiv) * cS2;

  const int lane = threadIdx.x & 31;
  const int wid  = threadIdx.x >> 5;
  const int l15  = lane & 15;
  const int hi8  = (lane >> 4) << 3;  // 0 or 8

  const int mBase = blockIdx.x * 128 + (wid & 1) * 64;
  const int nBase = blockIdx.y * 256 + (wid >> 1) * 64;

  const bf16* pa[4];
  const bf16* pw[4];
#pragma unroll
  for (int f = 0; f < 4; ++f) {
    pa[f] = A + (long long)min(mBase + f * 16 + l15, M - 1) * lda + hi8;
    pw[f] = W + (long long)min(nBase + f * 16 + l15, N - 1) * ldw + hi8;
  }

  const v8f zero = {0.f,0.f,0.f,0.f,0.f,0.f,0.f,0.f};
  v8f acc[4][4];
#pragma unroll
  for (int mi = 0; mi < 4; ++mi)
#pragma unroll
    for (int ni = 0; ni < 4; ++ni) acc[mi][ni] = zero;

  for (int kb = 0; kb < K; kb += 32) {
    v16bf a[4], b[4];
#pragma unroll
    for (int f = 0; f < 4; ++f) a[f] = load_frag(pa[f] + kb);
#pragma unroll
    for (int f = 0; f < 4; ++f) b[f] = load_frag(pw[f] + kb);
    if (kb + 64 <= K) {  // global_prefetch_b8 for the next K tile
#pragma unroll
      for (int f = 0; f < 4; ++f) {
        __builtin_prefetch(pa[f] + kb + 32, 0, 1);
        __builtin_prefetch(pw[f] + kb + 32, 0, 1);
      }
    }
#pragma unroll
    for (int mi = 0; mi < 4; ++mi)
#pragma unroll
      for (int ni = 0; ni < 4; ++ni)
        acc[mi][ni] = __builtin_amdgcn_wmma_f32_16x16x32_bf16(
            false, a[mi], false, b[ni], (short)0, acc[mi][ni], false, false);
  }

  const int bOuter = (mode == 3) ? (bidx / cNH) : 0;

#pragma unroll
  for (int mi = 0; mi < 4; ++mi) {
#pragma unroll
    for (int ni = 0; ni < 4; ++ni) {
      int col = nBase + ni * 16 + l15;
      if (col >= N) continue;
      float bv = (mode == 1 || mode == 2) ? bias[col] : 0.0f;
#pragma unroll
      for (int v = 0; v < 8; ++v) {
        int row = mBase + mi * 16 + hi8 + v;  // C: VGPR v -> M = v + (lane>=16?8:0)
        if (row >= M) continue;
        float x = acc[mi][ni][v];
        if (mode == 1) {
          x += bv;
        } else if (mode == 2) {
          x += bv;
          x = 0.5f * x * (1.0f + erff(x * 0.70710678118654752f));  // exact gelu
        } else if (mode == 3) {
          x *= scale;
          bool ok = (col < cVP) || (mask[(long long)bOuter * cTL + (col - cVP)] != 0);
          if (!ok) x = -3.402823466e38f;
        }
        long long idx = cOff + (long long)row * ldc + col;
        if (Ch) Ch[idx] = (bf16)x;
        else    Cf[idx] = x;
      }
    }
  }
}

// ---------------- layernorm( res + y ) -> fp32 + bf16 ----------------------
// One block (256 thr) per row of H=768. Row remap for concat placement.
__global__ __launch_bounds__(256)
void ln_res_kernel(const float* __restrict__ y, const float* __restrict__ res,
                   const float* __restrict__ g, const float* __restrict__ bb,
                   float* __restrict__ outF, bf16* __restrict__ outB,
                   int inRPB, int outRPB, int outOff)
{
  __shared__ float sm[9];
  const int r = blockIdx.x;
  const int orow = (r / inRPB) * outRPB + outOff + (r % inRPB);
  const float* yr = y + (long long)r * cH;
  const float* rr = res ? res + (long long)orow * cH : nullptr;

  float vals[3];
  float s = 0.f;
#pragma unroll
  for (int i = 0; i < 3; ++i) {
    int c = threadIdx.x + i * 256;
    float v = yr[c];
    if (rr) v += rr[c];
    vals[i] = v; s += v;
  }
  float mean = blockRedSum(s, sm) * (1.0f / cH);
  float s2 = 0.f;
#pragma unroll
  for (int i = 0; i < 3; ++i) { float d = vals[i] - mean; s2 += d * d; }
  float var = blockRedSum(s2, sm) * (1.0f / cH);
  float inv = rsqrtf(var + 1e-5f);
#pragma unroll
  for (int i = 0; i < 3; ++i) {
    int c = threadIdx.x + i * 256;
    float o = (vals[i] - mean) * inv * g[c] + bb[c];
    outF[(long long)orow * cH + c] = o;
    outB[(long long)orow * cH + c] = (bf16)o;
  }
}

// ---------------- softmax over S=512, write bf16 P -------------------------
__global__ __launch_bounds__(256)
void softmax_bf16_kernel(const float* __restrict__ sc, bf16* __restrict__ p)
{
  __shared__ float sm[9];
  const long long roff = (long long)blockIdx.x * cS;
  float v0 = sc[roff + threadIdx.x];
  float v1 = sc[roff + threadIdx.x + 256];
  float m = blockRedMax(fmaxf(v0, v1), sm);
  float e0 = expf(v0 - m), e1 = expf(v1 - m);
  float ssum = blockRedSum(e0 + e1, sm);
  float inv = 1.0f / ssum;
  p[roff + threadIdx.x]       = (bf16)(e0 * inv);
  p[roff + threadIdx.x + 256] = (bf16)(e1 * inv);
}

// ------- qkv [B,S,3H] bf16 -> Q,K [B*NH,S,HD], Vt [B*NH,HD,S] (bf16) -------
__global__ __launch_bounds__(256)
void extract_qkv_kernel(const bf16* __restrict__ qkv, bf16* __restrict__ q,
                        bf16* __restrict__ k, bf16* __restrict__ vt)
{
  long long idx = (long long)blockIdx.x * 256 + threadIdx.x;
  const long long total = (long long)cB * cS * cH;
  if (idx >= total) return;
  int d = (int)(idx % cHD);
  long long t = idx / cHD;
  int h = (int)(t % cNH); t /= cNH;
  int s = (int)(t % cS);
  int b = (int)(t / cS);
  const bf16* src = qkv + ((long long)b * cS + s) * (3 * cH) + h * cHD + d;
  long long bh = (long long)b * cNH + h;
  q [(bh * cS + s) * cHD + d] = src[0];
  k [(bh * cS + s) * cHD + d] = src[cH];
  vt[(bh * cHD + d) * cS + s] = src[2 * cH];
}

// ---------------- small elementwise helpers --------------------------------
__global__ __launch_bounds__(256)
void cvt_f32_bf16_kernel(const float* __restrict__ in, bf16* __restrict__ out, long long n)
{
  long long i = (long long)blockIdx.x * 256 + threadIdx.x;
  if (i < n) out[i] = (bf16)in[i];
}
__global__ __launch_bounds__(256)
void copy_f32_kernel(const float* __restrict__ in, float* __restrict__ out, long long n)
{
  long long i = (long long)blockIdx.x * 256 + threadIdx.x;
  if (i < n) out[i] = in[i];
}
// combined_mask (output 2): [B,S] -> 1.0 for vision / mask byte for text
__global__ __launch_bounds__(256)
void mask_out_kernel(const unsigned char* __restrict__ tmask, float* __restrict__ out)
{
  int i = blockIdx.x * 256 + threadIdx.x;
  if (i >= cB * cS) return;
  int b = i / cS, s = i % cS;
  out[i] = (s < cVP) ? 1.0f : (tmask[(long long)b * cTL + (s - cVP)] ? 1.0f : 0.0f);
}

// ---------------------------------------------------------------------------
extern "C" void kernel_launch(void* const* d_in, const int* in_sizes, int n_in,
                              void* d_out, int out_size, void* d_ws, size_t ws_size,
                              hipStream_t stream) {
  (void)in_sizes; (void)n_in; (void)out_size; (void)ws_size;

  // ----- inputs (setup_inputs order) -----
  const float* vision = (const float*)d_in[0];
  const float* text   = (const float*)d_in[1];
  const unsigned char* tmask = (const unsigned char*)d_in[2];  // jnp bool -> 1 byte
  const float* vp_w = (const float*)d_in[3];  const float* vp_b = (const float*)d_in[4];
  const float* vp_g = (const float*)d_in[5];  const float* vp_be = (const float*)d_in[6];
  const float* tp_w = (const float*)d_in[7];  const float* tp_b = (const float*)d_in[8];
  const float* tp_g = (const float*)d_in[9];  const float* tp_be = (const float*)d_in[10];
  const float* in_w = (const float*)d_in[11]; const float* in_b = (const float*)d_in[12];
  const float* out_w = (const float*)d_in[13]; const float* out_b = (const float*)d_in[14];
  const float* ln1_g = (const float*)d_in[15]; const float* ln1_b = (const float*)d_in[16];
  const float* ve_w1 = (const float*)d_in[17]; const float* ve_b1 = (const float*)d_in[18];
  const float* ve_w2 = (const float*)d_in[19]; const float* ve_b2 = (const float*)d_in[20];
  const float* le_w1 = (const float*)d_in[21]; const float* le_b1 = (const float*)d_in[22];
  const float* le_w2 = (const float*)d_in[23]; const float* le_b2 = (const float*)d_in[24];
  const float* ln2_g = (const float*)d_in[25]; const float* ln2_b = (const float*)d_in[26];
  float* outp = (float*)d_out;

  // ----- workspace carve-out -----
  char* wsp = (char*)d_ws;
  auto take = [&](size_t nbytes) -> char* {
    char* p = wsp; wsp += (nbytes + 255) & ~(size_t)255; return p;
  };
  const long long szVP = (long long)cH * cVD;
  const long long szTP = (long long)cH * cH;
  const long long stIN = 3LL * cH * cH;
  const long long stOUT = (long long)cH * cH;
  const long long stW  = (long long)cFF * cH;
  const long long szIN = cL * stIN, szOUT = cL * stOUT, szW1 = cL * stW;
  const long long nVF = (long long)cB * cVP * cVD;
  const long long nTF = (long long)cB * cTL * cH;
  const long long nX  = (long long)cB * cS * cH;      // 6291456
  const long long nQKV = (long long)cB * cS * 3 * cH;
  const long long nSC = (long long)cB * cNH * cS * cS;
  const long long nHH = (long long)cB * cS * cFF;

  bf16* wb_vp  = (bf16*)take(szVP * 2);
  bf16* wb_tp  = (bf16*)take(szTP * 2);
  bf16* wb_in  = (bf16*)take(szIN * 2);
  bf16* wb_out = (bf16*)take(szOUT * 2);
  bf16* wb_vw1 = (bf16*)take(szW1 * 2);
  bf16* wb_vw2 = (bf16*)take(szW1 * 2);
  bf16* wb_lw1 = (bf16*)take(szW1 * 2);
  bf16* wb_lw2 = (bf16*)take(szW1 * 2);
  bf16* vfb = (bf16*)take(nVF * 2);
  bf16* tfb = (bf16*)take(nTF * 2);
  float* xf = (float*)take(nX * 4);
  bf16*  xb = (bf16*)take(nX * 2);
  bf16*  qkvb = (bf16*)take(nQKV * 2);
  bf16* qb  = (bf16*)take(nX * 2);
  bf16* kbuf = (bf16*)take(nX * 2);
  bf16* vtb = (bf16*)take(nX * 2);
  float* scoresf = (float*)take(nSC * 4);
  bf16*  pb = (bf16*)take(nSC * 2);
  bf16*  aob = (bf16*)take(nX * 2);
  float* yf = (float*)take(nX * 4);    // also projection tmp
  bf16*  hb = (bf16*)take(nHH * 2);

  auto cvt = [&](const float* src, bf16* dst, long long n) {
    cvt_f32_bf16_kernel<<<(unsigned)((n + 255) / 256), 256, 0, stream>>>(src, dst, n);
  };
  auto gemm = [&](const bf16* A, const bf16* W, float* Cf, bf16* Ch,
                  int M, int N, int K, int lda, int ldw, int ldc, int batch,
                  long long sA, long long sW, int cDiv, long long cS1, long long cS2,
                  const float* bias, int mode, float scale,
                  const unsigned char* mk) {
    dim3 g((M + 127) / 128, (N + 255) / 256, batch);
    gemm_bf16_wmma<<<g, 256, 0, stream>>>(A, W, Cf, Ch, M, N, K, lda, ldw, ldc,
                                          sA, sW, cDiv, cS1, cS2, bias, mode, scale, mk);
  };
  auto ln = [&](const float* y, const float* res, const float* g, const float* b,
                int rows, int inRPB, int outRPB, int outOff) {
    ln_res_kernel<<<rows, 256, 0, stream>>>(y, res, g, b, xf, xb, inRPB, outRPB, outOff);
  };

  // ----- fp32 -> bf16 conversions (weights + raw inputs) -----
  cvt(vision, vfb, nVF);
  cvt(text,   tfb, nTF);
  cvt(vp_w, wb_vp, szVP);   cvt(tp_w, wb_tp, szTP);
  cvt(in_w, wb_in, szIN);   cvt(out_w, wb_out, szOUT);
  cvt(ve_w1, wb_vw1, szW1); cvt(ve_w2, wb_vw2, szW1);
  cvt(le_w1, wb_lw1, szW1); cvt(le_w2, wb_lw2, szW1);

  // ----- input projections + LN, concat into x[B,512,H] -----
  gemm(vfb, wb_vp, yf, nullptr, cB * cVP, cH, cVD, cVD, cVD, cH, 1,
       0, 0, 1, 0, 0, vp_b, 1, 1.0f, nullptr);
  ln(yf, nullptr, vp_g, vp_be, cB * cVP, cVP, cS, 0);
  gemm(tfb, wb_tp, yf, nullptr, cB * cTL, cH, cH, cH, cH, cH, 1,
       0, 0, 1, 0, 0, tp_b, 1, 1.0f, nullptr);
  ln(yf, nullptr, tp_g, tp_be, cB * cTL, cTL, cS, cVP);

  const float scale = 1.0f / sqrtf((float)cHD);
  const long long xHalf = (long long)cVP * cH;       // lang token offset in x
  const long long hHalf = (long long)cVP * cFF;

  for (int i = 0; i < cL; ++i) {
    // qkv = x @ in_w^T + in_b   (bf16 out, consumed by extract)
    gemm(xb, wb_in + i * stIN, nullptr, qkvb, cB * cS, 3 * cH, cH, cH, cH, 3 * cH, 1,
         0, 0, 1, 0, 0, in_b + i * 3 * cH, 1, 1.0f, nullptr);
    extract_qkv_kernel<<<(unsigned)((nX + 255) / 256), 256, 0, stream>>>(qkvb, qb, kbuf, vtb);

    // scores = Q K^T * scale + key-mask   (batch = B*NH, fp32 for softmax)
    gemm(qb, kbuf, scoresf, nullptr, cS, cS, cHD, cHD, cHD, cS, cB * cNH,
         (long long)cS * cHD, (long long)cS * cHD, 1, (long long)cS * cS, 0,
         nullptr, 3, scale, tmask);
    softmax_bf16_kernel<<<cB * cNH * cS, 256, 0, stream>>>(scoresf, pb);

    // ao[b,s,h*HD+d] = P @ Vt^T  (bf16 out; C base = b*S*H + h*HD via cDiv=NH)
    gemm(pb, vtb, nullptr, aob, cS, cHD, cS, cS, cS, cH, cB * cNH,
         (long long)cS * cS, (long long)cHD * cS,
         cNH, (long long)cS * cH, (long long)cHD, nullptr, 0, 1.0f, nullptr);

    // out-proj + residual LN1
    gemm(aob, wb_out + i * stOUT, yf, nullptr, cB * cS, cH, cH, cH, cH, cH, 1,
         0, 0, 1, 0, 0, out_b + i * cH, 1, 1.0f, nullptr);
    ln(yf, xf, ln1_g + i * cH, ln1_b + i * cH, cB * cS, cS, cS, 0);

    // modality-split FFN: gelu(x W1^T + b1) W2^T + b2  (hidden kept bf16)
    gemm(xb, wb_vw1 + i * stW, nullptr, hb, cVP, cFF, cH, cH, cH, cFF, cB,
         (long long)cS * cH, 0, 1, (long long)cS * cFF, 0,
         ve_b1 + i * cFF, 2, 1.0f, nullptr);
    gemm(xb + xHalf, wb_lw1 + i * stW, nullptr, hb + hHalf, cTL, cFF, cH, cH, cH, cFF, cB,
         (long long)cS * cH, 0, 1, (long long)cS * cFF, 0,
         le_b1 + i * cFF, 2, 1.0f, nullptr);
    gemm(hb, wb_vw2 + i * stW, yf, nullptr, cVP, cH, cFF, cFF, cFF, cH, cB,
         (long long)cS * cFF, 0, 1, (long long)cS * cH, 0,
         ve_b2 + i * cH, 1, 1.0f, nullptr);
    gemm(hb + hHalf, wb_lw2 + i * stW, yf + xHalf, nullptr, cTL, cH, cFF, cFF, cFF, cH, cB,
         (long long)cS * cFF, 0, 1, (long long)cS * cH, 0,
         le_b2 + i * cH, 1, 1.0f, nullptr);
    ln(yf, xf, ln2_g + i * cH, ln2_b + i * cH, cB * cS, cS, cS, 0);
  }

  // ----- outputs: x (fp32) then combined_mask (as float) -----
  copy_f32_kernel<<<(unsigned)((nX + 255) / 256), 256, 0, stream>>>(xf, outp, nX);
  mask_out_kernel<<<(cB * cS + 255) / 256, 256, 0, stream>>>(tmask, outp + nX);
}
